// neuron_layer_14946486190131
// MI455X (gfx1250) — matrix operand order repaired and checked by reading, not compiled, not run on hardware
//
#include <hip/hip_runtime.h>

// ---------------------------------------------------------------------------
// LIF spiking layer for MI455X (gfx1250, wave32, WMMA).
//   psp_t = x_t @ W.T + b ; v = v*decay + psp - r ; s = v>1 ; r = r*rdecay + s*rv
// Strategy: partition N across blocks (recurrence is per-neuron -> no global
// sync). 1 wave / 16-neuron tile, v/r state in VGPRs, W slice (64KB f16) in
// LDS, x_t fragments streamed from L2. GEMM via v_wmma_f32_16x16x32_f16.
// ---------------------------------------------------------------------------

typedef _Float16 half_t;
typedef __attribute__((ext_vector_type(16))) _Float16 v16h;
typedef __attribute__((ext_vector_type(8)))  float    v8f;

#define B_SZ    64
#define NIN     2048
#define NOUT    2048
#define T_STEPS 100
#define KT      (NIN / 32)   // 64 k-tiles of 32
#define MT      (B_SZ / 16)  // 4  m-tiles of 16 (batch)
#define NT      (NOUT / 16)  // 128 n-tiles of 16 (neurons)
#define FRAG_HALFS 512       // one 16x32 fragment: 32 lanes * 16 halfs

// ---- Pack W[n][k] (f32, row-major) into WMMA B-matrix (32x16 KxN) fragments.
// B layout (matches A): K = (j<4 ? 0 : 16) + (lane>=16 ? 8 : 0) + 2*(j&3) + {0,1};
// column n = lane%16. Stored 16 halfs/lane contiguous.
__global__ __launch_bounds__(32) void pack_W_kernel(const float* __restrict__ W,
                                                    half_t* __restrict__ Wh) {
  const int frag = blockIdx.x;           // nt*KT + kt
  const int kt   = frag & (KT - 1);
  const int nt   = frag >> 6;
  const int L    = threadIdx.x;
  const int n    = nt * 16 + (L & 15);
  const int kbase = kt * 32 + ((L >> 4) << 3);
  half_t* dst = Wh + (size_t)frag * FRAG_HALFS + L * 16;
  const float* src = W + (size_t)n * NIN;
#pragma unroll
  for (int j = 0; j < 8; ++j) {
    dst[2 * j]     = (half_t)src[kbase + ((j >> 2) << 4) + ((j & 3) << 1)];
    dst[2 * j + 1] = (half_t)src[kbase + ((j >> 2) << 4) + ((j & 3) << 1) + 1];
  }
}

// ---- Pack x[b][k][t] (f32, time innermost) into WMMA A-matrix (16x32 MxK)
// fragments per timestep. A layout (ISA 7.12.2, 16-bit A 16x32):
//   K = (j<4 ? 0 : 16) + (lane>=16 ? 8 : 0) + 2*(j&3) + h ; M = lane%16.
__global__ __launch_bounds__(32) void pack_x_kernel(const float* __restrict__ x,
                                                    half_t* __restrict__ xh) {
  const int frag = blockIdx.x;           // (t*MT + mt)*KT + kt
  const int kt   = frag & (KT - 1);
  const int mt   = (frag >> 6) & (MT - 1);
  const int t    = frag >> 8;
  const int L    = threadIdx.x;
  const int b    = mt * 16 + (L & 15);
  half_t* dst = xh + (size_t)frag * FRAG_HALFS + L * 16;
#pragma unroll
  for (int j = 0; j < 8; ++j) {
#pragma unroll
    for (int h = 0; h < 2; ++h) {
      const int k  = ((j >> 2) << 4) + ((L >> 4) << 3) + ((j & 3) << 1) + h;
      const int kk = kt * 32 + k;
      dst[2 * j + h] = (half_t)x[((size_t)b * NIN + kk) * T_STEPS + t];
    }
  }
}

// ---- Main recurrent kernel: one wave per 16-neuron tile.
__global__ __launch_bounds__(32) void lif_layer_kernel(
    const half_t* __restrict__ xh, const half_t* __restrict__ Wh,
    const float* __restrict__ bias, const float* __restrict__ decay_v,
    const float* __restrict__ reset_decay, const float* __restrict__ reset_v,
    float* __restrict__ out_spk, float* __restrict__ out_v,
    float* __restrict__ out_r) {
  __shared__ half_t ldsW[KT * FRAG_HALFS];  // 64 KB: this tile's W slice, reused 100x

  const int nt = blockIdx.x;
  const int L  = threadIdx.x;

  // Stage W fragments for this n-tile into LDS (int4 = 16B copies).
  {
    const int4* src = (const int4*)(Wh + (size_t)nt * KT * FRAG_HALFS);
    int4* dst = (int4*)ldsW;
    for (int i = L; i < (KT * FRAG_HALFS) / 8; i += 32) dst[i] = src[i];
  }
  __syncthreads();

  // Per-lane neuron parameters: C/D layout -> column n = lane%16.
  const int   n  = nt * 16 + (L & 15);
  const float bn = bias[n];
  const float dv = decay_v[n];
  const float rd = reset_decay[n];
  const float rv = reset_v[n];
  const int   mrow = (L >> 4) * 8;  // C/D layout: lanes 16-31 hold rows M=8..15

  float v[MT][8], r[MT][8];
#pragma unroll
  for (int mt = 0; mt < MT; ++mt)
#pragma unroll
    for (int e = 0; e < 8; ++e) { v[mt][e] = 0.0f; r[mt][e] = 0.0f; }

  const half_t* lane_x = xh + L * 16;

  for (int t = 0; t < T_STEPS; ++t) {
    if (t + 1 < T_STEPS)  // hint next timestep's A fragments toward the caches
      __builtin_prefetch(lane_x + (size_t)((t + 1) * MT) * KT * FRAG_HALFS, 0, 0);

    v8f acc[MT];
#pragma unroll
    for (int mt = 0; mt < MT; ++mt) acc[mt] = (v8f){0, 0, 0, 0, 0, 0, 0, 0};

    for (int kt = 0; kt < KT; ++kt) {
      const v16h bf = *(const v16h*)(&ldsW[kt * FRAG_HALFS + L * 16]);
#pragma unroll
      for (int mt = 0; mt < MT; ++mt) {
        const v16h af = *(const v16h*)(lane_x +
            (size_t)((t * MT + mt) * KT + kt) * FRAG_HALFS);
        acc[mt] = __builtin_amdgcn_wmma_f32_16x16x32_f16(
            /*neg_a=*/false, af, /*neg_b=*/false, bf,
            /*c_mod=*/(short)0, acc[mt], /*reuse_a=*/false, /*reuse_b=*/false);
      }
    }

    // LIF state update + spike emission.
#pragma unroll
    for (int mt = 0; mt < MT; ++mt) {
#pragma unroll
      for (int e = 0; e < 8; ++e) {
        const float psp = acc[mt][e] + bn;
        const float vv  = v[mt][e] * dv + psp - r[mt][e];
        const float s   = (vv > 1.0f) ? 1.0f : 0.0f;
        v[mt][e] = vv;
        r[mt][e] = r[mt][e] * rd + s * rv;
        const int b = mt * 16 + mrow + e;
        out_spk[((size_t)b * NOUT + n) * T_STEPS + t] = s;
      }
    }
  }

  // Final membrane / reset states.
#pragma unroll
  for (int mt = 0; mt < MT; ++mt) {
#pragma unroll
    for (int e = 0; e < 8; ++e) {
      const int b = mt * 16 + mrow + e;
      out_v[(size_t)b * NOUT + n] = v[mt][e];
      out_r[(size_t)b * NOUT + n] = r[mt][e];
    }
  }
}

extern "C" void kernel_launch(void* const* d_in, const int* in_sizes, int n_in,
                              void* d_out, int out_size, void* d_ws, size_t ws_size,
                              hipStream_t stream) {
  const float* x   = (const float*)d_in[0];  // [64, 2048, 100]
  const float* W   = (const float*)d_in[1];  // [2048, 2048]
  const float* b   = (const float*)d_in[2];  // [2048]
  const float* dv  = (const float*)d_in[3];
  const float* rdc = (const float*)d_in[4];
  const float* rv  = (const float*)d_in[5];

  half_t* xh = (half_t*)d_ws;                                   // ~25 MB
  half_t* Wh = xh + (size_t)T_STEPS * MT * KT * FRAG_HALFS;     // ~8 MB

  float* out_spk = (float*)d_out;                               // [64,2048,100]
  float* out_v   = out_spk + (size_t)B_SZ * NOUT * T_STEPS;     // [64,2048]
  float* out_r   = out_v + (size_t)B_SZ * NOUT;                 // [64,2048]

  pack_W_kernel<<<NT * KT, 32, 0, stream>>>(W, Wh);
  pack_x_kernel<<<T_STEPS * MT * KT, 32, 0, stream>>>(x, xh);
  lif_layer_kernel<<<NT, 32, 0, stream>>>(xh, Wh, b, dv, rdc, rv,
                                          out_spk, out_v, out_r);
}
